// MultiModalFusion_42125039239834
// MI455X (gfx1250) — compile-verified, matrix operand order checked by reading, not run on hardware
//
#include <hip/hip_runtime.h>

// ---------------------------------------------------------------------------
// MHA with grid bias for MI455X (gfx1250), wave32, WMMA f16 -> f32 accum.
//   kernel 1: qkv projection  (f32 in -> f16 q/k/v in (B,NH,S,HD) layout)
//   kernel 2: fused flash-style attention (f16 in/out, f32 softmax state)
//             K tile staged via GLOBAL_LOAD_ASYNC_TO_LDS_B128 (ASYNCcnt)
//   kernel 3: output projection (f16 in -> f32 out + bias), async X tile
// Workspace: 4 buffers of M*HID f16 = 32 MB.
// ---------------------------------------------------------------------------

typedef __attribute__((ext_vector_type(16))) _Float16 v16h;
typedef __attribute__((ext_vector_type(8)))  _Float16 half8;
typedef __attribute__((ext_vector_type(8)))  float    v8f;
typedef __attribute__((ext_vector_type(4)))  float    f4v;

#define B_   4
#define S_   1024
#define HID_ 1024
#define NH_  16
#define HD_  64
#define M_   (B_ * S_)

// 16-bit A-fragment gather (ISA 7.12.2): lane holds one row; k-chunks
// [kb, kb+7] and [16+kb, 16+kb+7] with kb = (lane<16) ? 0 : 8.
// B fragments use the mirror layout (lane = column), so x@W^T and q@k^T
// gather B exactly like A from row-major storage.
__device__ __forceinline__ v16h load_frag(const _Float16* __restrict__ row, int kb) {
  half8 lo = *(const half8*)(row + kb);
  half8 hi = *(const half8*)(row + 16 + kb);
  v16h a;
#pragma unroll
  for (int i = 0; i < 8; ++i) { a[i] = lo[i]; a[i + 8] = hi[i]; }
  return a;
}

__device__ __forceinline__ v8f wmma32(v16h a, v16h b, v8f c) {
  return __builtin_amdgcn_wmma_f32_16x16x32_f16(false, a, false, b, (short)0, c,
                                                false, false);
}

__device__ __forceinline__ half8 cvt8(f4v a, f4v b) {
  half8 h;
#pragma unroll
  for (int i = 0; i < 4; ++i) { h[i] = (_Float16)a[i]; h[4 + i] = (_Float16)b[i]; }
  return h;
}

// Async global->LDS copy of 64 contiguous bytes (4 x B128). GVS addressing:
//   LDS[lds_off + ioff] = MEM[saddr + goff + ioff]; tracked by ASYNCcnt.
__device__ __forceinline__ void async_copy64(unsigned lds_off, unsigned goff,
                                             const void* saddr) {
  asm volatile(
      "global_load_async_to_lds_b128 %0, %1, %2\n\t"
      "global_load_async_to_lds_b128 %0, %1, %2 offset:16\n\t"
      "global_load_async_to_lds_b128 %0, %1, %2 offset:32\n\t"
      "global_load_async_to_lds_b128 %0, %1, %2 offset:48"
      :: "v"(lds_off), "v"(goff), "s"(saddr) : "memory");
}

__device__ __forceinline__ void async_copy32(unsigned lds_off, unsigned goff,
                                             const void* saddr) {
  asm volatile(
      "global_load_async_to_lds_b128 %0, %1, %2\n\t"
      "global_load_async_to_lds_b128 %0, %1, %2 offset:16"
      :: "v"(lds_off), "v"(goff), "s"(saddr) : "memory");
}

__device__ __forceinline__ void wait_async0() {
  asm volatile("s_wait_asynccnt 0x0" ::: "memory");
}

__device__ __forceinline__ unsigned lds_off32(const void* p) {
  return (unsigned)(size_t)p;  // flat LDS aperture: offset in addr[31:0]
}

// ---------------------------------------------------------------------------
// Kernel 1: q/k/v = X @ W^T + b, output f16 in (B,NH,S,HD) layout.
// Grid: (HID/64, M/128, 3). Block: 256 (8 waves; 4x2 wave grid, 32x32/wave).
// ---------------------------------------------------------------------------
__global__ __launch_bounds__(256)
void qkv_kernel(const float* __restrict__ Xq, const float* __restrict__ Xk,
                const float* __restrict__ Xv, const float* __restrict__ Wq,
                const float* __restrict__ Wk, const float* __restrict__ Wv,
                const float* __restrict__ bq, const float* __restrict__ bk,
                const float* __restrict__ bv, _Float16* __restrict__ Oq,
                _Float16* __restrict__ Ok, _Float16* __restrict__ Ov) {
  const int z = blockIdx.z;
  const float* X    = (z == 0) ? Xq : (z == 1) ? Xk : Xv;
  const float* W    = (z == 0) ? Wq : (z == 1) ? Wk : Wv;
  const float* bias = (z == 0) ? bq : (z == 1) ? bk : bv;
  _Float16*    O    = (z == 0) ? Oq : (z == 1) ? Ok : Ov;

  __shared__ __align__(16) _Float16 Xs[128][40];
  __shared__ __align__(16) _Float16 Ws[64][40];

  const int t = threadIdx.x;
  const int lane = t & 31, wave = t >> 5;
  const int wm = wave >> 1, wn = wave & 1;
  const int m0 = blockIdx.y * 128, n0 = blockIdx.x * 64;
  const int kb = (lane < 16) ? 0 : 8;
  const int lr = lane & 15, hi = lane >> 4;

  const int xrow = t >> 1, xhalf = t & 1;
  const int wrow = t >> 2, wq = t & 3;

  v8f acc[2][2] = {};

  for (int k0 = 0; k0 < HID_; k0 += 32) {
    __syncthreads();
    {  // X tile: 128 rows x 32 k (f32 -> f16)
      const float* g = X + (size_t)(m0 + xrow) * HID_ + k0 + xhalf * 16;
      f4v x0 = *(const f4v*)g;       f4v x1 = *(const f4v*)(g + 4);
      f4v x2 = *(const f4v*)(g + 8); f4v x3 = *(const f4v*)(g + 12);
      *(half8*)&Xs[xrow][xhalf * 16]     = cvt8(x0, x1);
      *(half8*)&Xs[xrow][xhalf * 16 + 8] = cvt8(x2, x3);
      if (k0 + 32 < HID_) __builtin_prefetch(g + 32, 0, 0);  // next k-step
    }
    {  // W tile: 64 rows x 32 k
      const float* g = W + (size_t)(n0 + wrow) * HID_ + k0 + wq * 8;
      f4v x0 = *(const f4v*)g; f4v x1 = *(const f4v*)(g + 4);
      *(half8*)&Ws[wrow][wq * 8] = cvt8(x0, x1);
      if (k0 + 32 < HID_) __builtin_prefetch(g + 32, 0, 0);
    }
    __syncthreads();

    v16h a[2], bf[2];
#pragma unroll
    for (int tm = 0; tm < 2; ++tm) a[tm]  = load_frag(&Xs[wm * 32 + tm * 16 + lr][0], kb);
#pragma unroll
    for (int tn = 0; tn < 2; ++tn) bf[tn] = load_frag(&Ws[wn * 32 + tn * 16 + lr][0], kb);
#pragma unroll
    for (int tm = 0; tm < 2; ++tm)
#pragma unroll
      for (int tn = 0; tn < 2; ++tn) acc[tm][tn] = wmma32(a[tm], bf[tn], acc[tm][tn]);
  }

  // Epilogue: bias + scatter into (B,NH,S,HD) f16.
#pragma unroll
  for (int tn = 0; tn < 2; ++tn) {
    const int n = n0 + wn * 32 + tn * 16 + lr;
    const float bn = bias[n];
    const int h = n >> 6, d = n & 63;
#pragma unroll
    for (int tm = 0; tm < 2; ++tm) {
#pragma unroll
      for (int r = 0; r < 8; ++r) {
        const int m = m0 + wm * 32 + tm * 16 + r + 8 * hi;
        const int bb = m >> 10, sq = m & (S_ - 1);
        O[(((size_t)(bb * NH_ + h) * S_) + sq) * HD_ + d] =
            (_Float16)(acc[tm][tn][r] + bn);
      }
    }
  }
}

// ---------------------------------------------------------------------------
// Kernel 2: fused attention. Grid: (S/64, B*NH). Block: 128 (4 waves).
// Each wave owns 16 query rows; streams 64-key tiles with online softmax.
// ---------------------------------------------------------------------------
__global__ __launch_bounds__(128)
void attn_kernel(const _Float16* __restrict__ Q, const _Float16* __restrict__ K,
                 const _Float16* __restrict__ V, _Float16* __restrict__ Oa) {
  __shared__ __align__(16) _Float16 Ks[64][72];
  __shared__ __align__(16) _Float16 Vs[64][72];      // transposed: [d][key]
  __shared__ __align__(16) _Float16 Ps[4][16][72];   // per-wave P staging

  const int t = threadIdx.x, lane = t & 31, wave = t >> 5;
  const int lr = lane & 15, hi = lane >> 4;
  const int kb = (lane < 16) ? 0 : 8;
  const int bh = blockIdx.y;                 // b*NH + h
  const int iq_base = blockIdx.x * 64 + wave * 16;

  const _Float16* qbase = Q + (size_t)bh * S_ * HD_;
  const _Float16* kbase = K + (size_t)bh * S_ * HD_;
  const _Float16* vbase = V + (size_t)bh * S_ * HD_;

  // Q fragments (16 rows x 64 d) held in registers for the whole kernel.
  v16h aq[2];
  {
    const _Float16* qr = qbase + (size_t)(iq_base + lr) * HD_;
    aq[0] = load_frag(qr, kb);
    aq[1] = load_frag(qr + 32, kb);
  }

  float mrow[8], lrow[8];
  v8f o[4] = {};
#pragma unroll
  for (int r = 0; r < 8; ++r) { mrow[r] = -1e30f; lrow[r] = 0.0f; }
  const float scale = 0.125f;  // 1/sqrt(64)

  const int row = t >> 1, half = t & 1;
  const unsigned ks_lds = lds_off32(&Ks[row][half * 32]);

  for (int kt = 0; kt < S_; kt += 64) {
    __syncthreads();  // previous iteration's tile reads complete
    {
      // K tile: async global->LDS (4 x B128 per thread, ASYNCcnt-tracked).
      const unsigned goff =
          (unsigned)(((kt + row) * HD_ + half * 32) * sizeof(_Float16));
      async_copy64(ks_lds, goff, (const void*)kbase);

      // V tile: VGPR path with transpose into Vs[d][key].
      const half8* gv = (const half8*)(vbase + (size_t)(kt + row) * HD_ + half * 32);
      half8 v0 = gv[0], v1 = gv[1], v2 = gv[2], v3 = gv[3];
#pragma unroll
      for (int j = 0; j < 8; ++j) {
        Vs[half * 32 + j][row]      = v0[j];
        Vs[half * 32 + 8 + j][row]  = v1[j];
        Vs[half * 32 + 16 + j][row] = v2[j];
        Vs[half * 32 + 24 + j][row] = v3[j];
      }
      if (kt + 64 < S_)  // prefetch next V tile (global_prefetch_b8)
        __builtin_prefetch(vbase + (size_t)(kt + 64 + row) * HD_ + half * 32, 0, 0);
    }
    wait_async0();    // K tile async writes landed in LDS
    __syncthreads();  // tiles visible to all waves

    // Scores: 4 key sub-tiles of 16; each is two chained k=32 WMMAs.
    v8f s[4];
#pragma unroll
    for (int j = 0; j < 4; ++j) {
      v16h bk0 = load_frag(&Ks[j * 16 + lr][0], kb);
      v16h bk1 = load_frag(&Ks[j * 16 + lr][32], kb);
      v8f c = {};
      c = wmma32(aq[0], bk0, c);
      c = wmma32(aq[1], bk1, c);
      s[j] = c;
    }

    // Scale + grid bias + row max.
    float rmax[8];
#pragma unroll
    for (int r = 0; r < 8; ++r) rmax[r] = -1e30f;
#pragma unroll
    for (int j = 0; j < 4; ++j) {
      const int jg = kt + j * 16 + lr;
      const int jgc = jg >> 5, jgm = jg & 31;
#pragma unroll
      for (int r = 0; r < 8; ++r) {
        const int i = iq_base + r + 8 * hi;
        const float bias = (((i >> 5) == jgc) || ((i & 31) == jgm)) ? 0.1f : 0.0f;
        const float val = s[j][r] * scale + bias;
        s[j][r] = val;
        rmax[r] = fmaxf(rmax[r], val);
      }
    }
#pragma unroll
    for (int off = 1; off < 16; off <<= 1)
#pragma unroll
      for (int r = 0; r < 8; ++r)
        rmax[r] = fmaxf(rmax[r], __shfl_xor(rmax[r], off, 32));

    // Online softmax update.
    float alpha[8], rsum[8];
#pragma unroll
    for (int r = 0; r < 8; ++r) {
      const float mn = fmaxf(mrow[r], rmax[r]);
      alpha[r] = __expf(mrow[r] - mn);
      mrow[r] = mn;
      rsum[r] = 0.0f;
    }
#pragma unroll
    for (int j = 0; j < 4; ++j)
#pragma unroll
      for (int r = 0; r < 8; ++r) {
        const float p = __expf(s[j][r] - mrow[r]);
        s[j][r] = p;
        rsum[r] += p;
      }
#pragma unroll
    for (int off = 1; off < 16; off <<= 1)
#pragma unroll
      for (int r = 0; r < 8; ++r) rsum[r] += __shfl_xor(rsum[r], off, 32);
#pragma unroll
    for (int r = 0; r < 8; ++r) lrow[r] = lrow[r] * alpha[r] + rsum[r];
#pragma unroll
    for (int dd = 0; dd < 4; ++dd)
#pragma unroll
      for (int r = 0; r < 8; ++r) o[dd][r] *= alpha[r];

    // P (f16) -> LDS, re-read in A-fragment layout.
#pragma unroll
    for (int j = 0; j < 4; ++j)
#pragma unroll
      for (int r = 0; r < 8; ++r)
        Ps[wave][r + 8 * hi][j * 16 + lr] = (_Float16)s[j][r];
    __syncthreads();  // orders Ps write -> read (and DS counters)

    v16h ap0 = load_frag(&Ps[wave][lr][0], kb);
    v16h ap1 = load_frag(&Ps[wave][lr][32], kb);
#pragma unroll
    for (int dd = 0; dd < 4; ++dd) {
      v16h bv0 = load_frag(&Vs[dd * 16 + lr][0], kb);
      v16h bv1 = load_frag(&Vs[dd * 16 + lr][32], kb);
      o[dd] = wmma32(ap0, bv0, o[dd]);
      o[dd] = wmma32(ap1, bv1, o[dd]);
    }
  }

  // Normalize and store f16 in (B,S,HID) layout for the output projection.
  const int bb = bh / NH_, h = bh % NH_;
#pragma unroll
  for (int dd = 0; dd < 4; ++dd)
#pragma unroll
    for (int r = 0; r < 8; ++r) {
      const int sq = iq_base + r + 8 * hi;
      const float val = o[dd][r] / lrow[r];
      Oa[((size_t)(bb * S_ + sq)) * HID_ + h * HD_ + dd * 16 + lr] = (_Float16)val;
    }
}

// ---------------------------------------------------------------------------
// Kernel 3: out = A @ Wo^T + bo, f16 in -> f32 out.
// Grid: (HID/64, M/128). Block: 256.
// ---------------------------------------------------------------------------
__global__ __launch_bounds__(256)
void oproj_kernel(const _Float16* __restrict__ X, const float* __restrict__ W,
                  const float* __restrict__ bias, float* __restrict__ Out) {
  __shared__ __align__(16) _Float16 Xs[128][40];
  __shared__ __align__(16) _Float16 Ws[64][40];

  const int t = threadIdx.x;
  const int lane = t & 31, wave = t >> 5;
  const int wm = wave >> 1, wn = wave & 1;
  const int m0 = blockIdx.y * 128, n0 = blockIdx.x * 64;
  const int kb = (lane < 16) ? 0 : 8;
  const int lr = lane & 15, hi = lane >> 4;

  const int xrow = t >> 1, xhalf = t & 1;
  const int wrow = t >> 2, wq = t & 3;
  const unsigned xs_lds = lds_off32(&Xs[xrow][xhalf * 16]);

  v8f acc[2][2] = {};

  for (int k0 = 0; k0 < HID_; k0 += 32) {
    __syncthreads();
    {  // X tile: 128 rows x 32 k (f16), async global->LDS (2 x B128/thread)
      const unsigned goff = (unsigned)(((size_t)(m0 + xrow) * HID_ + k0 +
                                        xhalf * 16) * sizeof(_Float16));
      async_copy32(xs_lds, goff, (const void*)X);
    }
    {  // W tile: 64 rows x 32 k (f32 -> f16)
      const float* g = W + (size_t)(n0 + wrow) * HID_ + k0 + wq * 8;
      f4v x0 = *(const f4v*)g; f4v x1 = *(const f4v*)(g + 4);
      *(half8*)&Ws[wrow][wq * 8] = cvt8(x0, x1);
      if (k0 + 32 < HID_) __builtin_prefetch(g + 32, 0, 0);
    }
    wait_async0();
    __syncthreads();

    v16h a[2], bf[2];
#pragma unroll
    for (int tm = 0; tm < 2; ++tm) a[tm]  = load_frag(&Xs[wm * 32 + tm * 16 + lr][0], kb);
#pragma unroll
    for (int tn = 0; tn < 2; ++tn) bf[tn] = load_frag(&Ws[wn * 32 + tn * 16 + lr][0], kb);
#pragma unroll
    for (int tm = 0; tm < 2; ++tm)
#pragma unroll
      for (int tn = 0; tn < 2; ++tn) acc[tm][tn] = wmma32(a[tm], bf[tn], acc[tm][tn]);
  }

#pragma unroll
  for (int tn = 0; tn < 2; ++tn) {
    const int n = n0 + wn * 32 + tn * 16 + lr;
    const float bn = bias[n];
#pragma unroll
    for (int tm = 0; tm < 2; ++tm) {
#pragma unroll
      for (int r = 0; r < 8; ++r) {
        const int m = m0 + wm * 32 + tm * 16 + r + 8 * hi;
        Out[(size_t)m * HID_ + n] = acc[tm][tn][r] + bn;  // lanes 0..15 coalesce
      }
    }
  }
}

// ---------------------------------------------------------------------------
extern "C" void kernel_launch(void* const* d_in, const int* in_sizes, int n_in,
                              void* d_out, int out_size, void* d_ws, size_t ws_size,
                              hipStream_t stream) {
  (void)in_sizes; (void)n_in; (void)out_size; (void)ws_size;
  const float* query = (const float*)d_in[0];
  const float* key   = (const float*)d_in[1];
  const float* value = (const float*)d_in[2];
  const float* Wq = (const float*)d_in[3];
  const float* bq = (const float*)d_in[4];
  const float* Wk = (const float*)d_in[5];
  const float* bk = (const float*)d_in[6];
  const float* Wv = (const float*)d_in[7];
  const float* bv = (const float*)d_in[8];
  const float* Wo = (const float*)d_in[9];
  const float* bo = (const float*)d_in[10];
  float* out = (float*)d_out;

  // Workspace: 4 x (M*HID) f16 = 32 MB.
  _Float16* qws = (_Float16*)d_ws;
  _Float16* kws = qws + (size_t)M_ * HID_;
  _Float16* vws = kws + (size_t)M_ * HID_;
  _Float16* aws = vws + (size_t)M_ * HID_;

  qkv_kernel<<<dim3(HID_ / 64, M_ / 128, 3), 256, 0, stream>>>(
      query, key, value, Wq, Wk, Wv, bq, bk, bv, qws, kws, vws);
  attn_kernel<<<dim3(S_ / 64, B_ * NH_), 128, 0, stream>>>(qws, kws, vws, aws);
  oproj_kernel<<<dim3(HID_ / 64, M_ / 128), 256, 0, stream>>>(aws, Wo, bo, out);
}
